// CustomConv2d_36283883717253
// MI455X (gfx1250) — matrix operand order repaired and checked by reading, not compiled, not run on hardware
//
#include <hip/hip_runtime.h>

// MI455X / gfx1250 implicit-GEMM conv3x3 via V_WMMA_F32_16X16X32_BF16 with
// GLOBAL_LOAD_ASYNC_TO_LDS_B128 double-buffered staging.
// GEMM view: M=Cout=256, N=B*H*W=100352, K=Cin*9=1152.

typedef __bf16 bf16_t;
typedef __attribute__((ext_vector_type(8)))  __bf16 v8bf;
typedef __attribute__((ext_vector_type(16))) __bf16 v16bf;
typedef __attribute__((ext_vector_type(8)))  float  v8f;
typedef __attribute__((address_space(3))) char lds_char;

namespace {
constexpr int kCin    = 128;
constexpr int kCout   = 256;
constexpr int kHW     = 56;
constexpr int kNPix   = kHW * kHW;           // 3136
constexpr int kBatch  = 32;
constexpr int kKChunks = 9 * (kCin / 32);    // 36 K-chunks of 32
}

// byte offset of a shared-memory pointer within the workgroup's LDS allocation
__device__ __forceinline__ unsigned lds_off_of(const void* p) {
  return (unsigned)(unsigned long long)(const lds_char*)p;
}

// async DMA: 16 bytes per lane, global(saddr + voffset) -> LDS[ldsb]
__device__ __forceinline__ void async_copy16(unsigned ldsb, unsigned goff,
                                             const void* base) {
  asm volatile("global_load_async_to_lds_b128 %0, %1, %2"
               :: "v"(ldsb), "v"(goff), "s"(base)
               : "memory");
}

// ---- pre-pass: x NCHW fp32 -> NHWC bf16 (channels contiguous per pixel) ----
__global__ void cvt_x_nhwc_bf16(const float* __restrict__ x,
                                bf16_t* __restrict__ xn) {
  const int t   = blockIdx.x * 256 + threadIdx.x;   // one thread per pixel
  const int b   = t / kNPix;
  const int phw = t - b * kNPix;
  const float* src = x + (size_t)b * kCin * kNPix + phw;   // coalesced reads
  bf16_t* dst = xn + (size_t)t * kCin;
#pragma unroll 8
  for (int ci = 0; ci < kCin; ++ci)
    dst[ci] = (bf16_t)__builtin_nontemporal_load(&src[(size_t)ci * kNPix]);
}

// ---- pre-pass: weight OIHW fp32 -> [tap][Cout][Cin] bf16 ----
__global__ void cvt_w_bf16(const float* __restrict__ w,
                           bf16_t* __restrict__ wt) {
  const int d   = blockIdx.x * 256 + threadIdx.x;   // 0..294911
  const int ci  = d & 127;
  const int co  = (d >> 7) & 255;
  const int tap = d >> 15;                          // 256*128 = 2^15
  wt[d] = (bf16_t)__builtin_nontemporal_load(&w[(co * kCin + ci) * 9 + tap]);
}

// ---- main: 128(Cout) x 64(pixels) block tile, 8 waves, 4 WMMA accs/wave ----
__global__ __launch_bounds__(256)
void conv3x3_wmma_bf16(const bf16_t* __restrict__ xn,
                       const bf16_t* __restrict__ wt,
                       const float* __restrict__ bias,
                       float* __restrict__ out) {
  // [0,16384): A double buffer (2 x 128 rows x 64B)
  // [16384,24576): B double buffer (2 x 64 rows x 64B)
  // [24576,24592): dummy sink for OOB async lanes
  __shared__ alignas(64) char smem[2 * 8192 + 2 * 4096 + 16];

  const int t      = threadIdx.x;
  const int lane   = t & 31;
  const int wave   = t >> 5;
  const int wm     = wave & 3;      // cout sub-group: rows wm*32..+31
  const int wn     = wave >> 2;     // pixel sub-group: cols wn*32..+31
  const int lane16 = lane & 15;
  const int hi     = lane >> 4;

  const int pBase = blockIdx.x * 64;    // pixel tile within image
  const int mBase = blockIdx.y * 128;   // cout tile
  const int b     = blockIdx.z;

  // staging assignment: 16-byte segments
  const int am   = t >> 2;   // A rows am and am+64
  const int aseg = t & 3;
  const int bn   = t >> 2;   // B pixel row
  const int bseg = t & 3;
  const int p    = pBase + bn;
  const int oh   = p / kHW;
  const int ow   = p - oh * kHW;

  const unsigned smemBase = lds_off_of(smem);
  const unsigned aLds0 = smemBase + am * 64 + aseg * 16;           // + buf*8192
  const unsigned aLds1 = aLds0 + 64 * 64;
  const unsigned bLdsS = smemBase + 16384 + bn * 64 + bseg * 16;   // + buf*4096
  const unsigned dummyLds = smemBase + 24576;

  v8f acc[2][2];
  for (int i = 0; i < 2; ++i)
    for (int j = 0; j < 2; ++j)
      acc[i][j] = (v8f)0.0f;

  // issue the 3 async b128 staging ops for K-chunk c into buffer `buf`.
  // Exactly 3 per thread, unconditional -> uniform ASYNCcnt per wave.
  auto stage = [&](int c, int buf) {
    const int tap = c >> 2;
    const int kc  = (c & 3) << 5;
    const unsigned aGoff =
        (unsigned)(((tap << 8) + mBase + am) * kCin + kc) * 2 + aseg * 16;
    async_copy16(aLds0 + buf * 8192, aGoff, wt);
    async_copy16(aLds1 + buf * 8192, aGoff + 64u * kCin * 2, wt);

    const int r  = tap / 3;
    const int s  = tap - r * 3;
    const int ih = oh + r - 1;
    const int iw = ow + s - 1;
    const bool inb = ((unsigned)ih < (unsigned)kHW) &
                     ((unsigned)iw < (unsigned)kHW);
    // OOB lanes: redirect DMA to the dummy slot with a clamped valid address,
    // and write zeros to the real slot (tracked ds_store, flushed at barrier).
    const int ihc = inb ? ih : oh;
    const int iwc = inb ? iw : ow;
    const unsigned bGoff =
        (unsigned)((b * kNPix + ihc * kHW + iwc) * kCin + kc) * 2 + bseg * 16;
    const unsigned bLds = inb ? (bLdsS + buf * 4096) : dummyLds;
    async_copy16(bLds, bGoff, xn);
    if (!inb)
      *(uint4*)(smem + 16384 + buf * 4096 + bn * 64 + bseg * 16) =
          make_uint4(0u, 0u, 0u, 0u);
  };

  stage(0, 0);

#pragma unroll 2
  for (int c = 0; c < kKChunks; ++c) {
    const int buf = c & 1;
    if (c + 1 < kKChunks) {
      stage(c + 1, buf ^ 1);
      asm volatile("s_wait_asynccnt 0x3" ::: "memory");  // chunk c landed
    } else {
      asm volatile("s_wait_asynccnt 0x0" ::: "memory");
    }
    __syncthreads();   // all waves' chunk-c data visible

    // A fragments (16x32 bf16): lanes 0-15 hold K0-15 of row lane16,
    // lanes 16-31 hold K16-31 (same k order as B).
    const char* aP = smem + buf * 8192 + (wm * 32 + lane16) * 64 + hi * 32;
    v8bf a0lo = *(const v8bf*)(aP);
    v8bf a0hi = *(const v8bf*)(aP + 16);
    v8bf a1lo = *(const v8bf*)(aP + 1024);
    v8bf a1hi = *(const v8bf*)(aP + 1024 + 16);
    v16bf a0 = __builtin_shufflevector(a0lo, a0hi,
        0,1,2,3,4,5,6,7,8,9,10,11,12,13,14,15);
    v16bf a1 = __builtin_shufflevector(a1lo, a1hi,
        0,1,2,3,4,5,6,7,8,9,10,11,12,13,14,15);

    // B fragments (32x16 bf16): lanes 0-15 = K0-15 of column lane16,
    // lanes 16-31 = K16-31; pixel-major LDS tile makes this contiguous.
    const char* bP =
        smem + 16384 + buf * 4096 + (wn * 32 + lane16) * 64 + hi * 32;
    v16bf b0 = *(const v16bf*)(bP);
    v16bf b1 = *(const v16bf*)(bP + 1024);

    acc[0][0] = __builtin_amdgcn_wmma_f32_16x16x32_bf16(
        false, a0, false, b0, (short)0, acc[0][0], false, false);
    acc[0][1] = __builtin_amdgcn_wmma_f32_16x16x32_bf16(
        false, a0, false, b1, (short)0, acc[0][1], false, false);
    acc[1][0] = __builtin_amdgcn_wmma_f32_16x16x32_bf16(
        false, a1, false, b0, (short)0, acc[1][0], false, false);
    acc[1][1] = __builtin_amdgcn_wmma_f32_16x16x32_bf16(
        false, a1, false, b1, (short)0, acc[1][1], false, false);

    __syncthreads();   // chunk-c compute done; buffer reusable next iteration
  }

  // epilogue: C/D layout = VGPR j -> M = j (lanes 0-15) / 8+j (lanes 16-31),
  // N = lane16. Add bias, non-temporal fp32 NCHW stores (write-once stream).
  for (int mi = 0; mi < 2; ++mi) {
    const int coB = mBase + wm * 32 + mi * 16 + hi * 8;
    for (int ni = 0; ni < 2; ++ni) {
      const int pp = pBase + wn * 32 + ni * 16 + lane16;
      float* op = out + ((size_t)b * kCout + coB) * kNPix + pp;
#pragma unroll
      for (int j = 0; j < 8; ++j)
        __builtin_nontemporal_store(acc[mi][ni][j] + bias[coB + j],
                                    op + (size_t)j * kNPix);
    }
  }
}

extern "C" void kernel_launch(void* const* d_in, const int* in_sizes, int n_in,
                              void* d_out, int out_size, void* d_ws, size_t ws_size,
                              hipStream_t stream) {
  (void)in_sizes; (void)n_in; (void)out_size; (void)ws_size;
  const float* x      = (const float*)d_in[0];
  const float* weight = (const float*)d_in[1];
  const float* bias   = (const float*)d_in[2];
  // d_in[3] = approximate: controls HW approx level only; exact math ignores it.
  float* out = (float*)d_out;

  // workspace: [0, 25.7MB) x bf16 NHWC ; then 0.59MB weights bf16
  bf16_t* xnp = (bf16_t*)d_ws;
  bf16_t* wtp = xnp + (size_t)kBatch * kNPix * kCin;

  cvt_x_nhwc_bf16<<<(kBatch * kNPix) / 256, 256, 0, stream>>>(x, xnp);
  cvt_w_bf16<<<(9 * kCout * kCin) / 256, 256, 0, stream>>>(weight, wtp);

  dim3 grid(kNPix / 64, kCout / 128, kBatch);   // (49, 2, 32)
  conv3x3_wmma_bf16<<<grid, 256, 0, stream>>>(xnp, wtp, bias, out);
}
